// VQ_VAE_Model_62697932587087
// MI455X (gfx1250) — matrix operand order repaired and checked by reading, not compiled, not run on hardware
//
#include <hip/hip_runtime.h>
#include <hip/hip_fp16.h>

// ---------------------------------------------------------------------------
// VQ-VAE forward for MI455X (gfx1250, wave32, WMMA).
// ~40 GFLOP vs ~230 MB traffic: balanced at 23.3 TB/s only if conv2, the
// linears, and conv_t1 run on v_wmma_f32_16x16x32_f16. Degenerate convs
// (C_in=3 / C_out=3) stay on VALU. f16 intermediates halve activation BW.
// B-tiles are staged in LDS in WMMA-operand order (1 ds_load_b128 pair per
// operand); A-operands come from pre-shuffled weights (b128 loads) or from
// two contiguous 16B runs per the ISA 16-bit A layout.
// ---------------------------------------------------------------------------

typedef __attribute__((ext_vector_type(16))) _Float16 v16h;
typedef __attribute__((ext_vector_type(8)))  _Float16 v8h;
typedef __attribute__((ext_vector_type(8)))  float    v8f;

__device__ __forceinline__ v8f wmma16(v16h a, v16h b, v8f c) {
  return __builtin_amdgcn_wmma_f32_16x16x32_f16(false, a, false, b, (short)0, c,
                                                false, false);
}

// ISA 7.12.2 16-bit A 16x32: lane holds row M=lane&15; K elements are two
// contiguous 8-runs: [k0 + half*8, +7] and [k0 + 16 + half*8, +7].
__device__ __forceinline__ int a_k_of(int lane, int j) {
  int h = (lane >> 4) & 1, v = j >> 1, lo = j & 1;
  return (v < 4) ? (h * 8 + v * 2 + lo) : (16 + h * 8 + (v - 4) * 2 + lo);
}

// Build A operand from two contiguous 16B runs (two global_load_b128).
__device__ __forceinline__ v16h load_a_runs(const _Float16* __restrict__ base) {
  v8h lo = *(const v8h*)(base);        // k0+half*8 .. +7
  v8h hi = *(const v8h*)(base + 16);   // k0+16+half*8 .. +7
  return __builtin_shufflevector(lo, hi, 0, 1, 2, 3, 4, 5, 6, 7,
                                 8, 9, 10, 11, 12, 13, 14, 15);
}
// B 32x16: lane holds col N=lane&15, K = (j>>3)*16 + half*8 + (j&7) (same order as A).

#define BATCH 256
#define LATENT 64
#define NCODES 512
#define FLAT 32768   // 128*16*16

// ====================== weight pre-shuffle kernels =========================
// conv2 weights [128,64,4,4] -> A-operand order [mt(8)][kt(32)][lane(32)][16]
__global__ void prep_w2(const float* __restrict__ w, _Float16* __restrict__ o) {
  int i = blockIdx.x * 256 + threadIdx.x;           // 131072
  int j = i & 15, lane = (i >> 4) & 31, kt = (i >> 9) & 31, mt = i >> 14;
  int k = a_k_of(lane, j) + 32 * kt;
  int oc = mt * 16 + (lane & 15);
  o[i] = (_Float16)w[oc * 1024 + k];
}

// enc_wl [32768,64] -> B-operand order [nt(4)][kt(1024)][lane(32)][16]
__global__ void prep_encwl(const float* __restrict__ w, _Float16* __restrict__ o) {
  int i = blockIdx.x * 256 + threadIdx.x;           // 2097152
  int j = i & 15, lane = (i >> 4) & 31, kt = (i >> 9) & 1023, nt = i >> 19;
  int k = 32 * kt + ((j >> 3) << 4) + ((lane >> 4) & 1) * 8 + (j & 7);
  int n = nt * 16 + (lane & 15);
  o[i] = (_Float16)w[k * 64 + n];
}

// dec_wl [64,32768] -> B-operand order [nt(2048)][kt(2)][lane(32)][16]
__global__ void prep_decwl(const float* __restrict__ w, _Float16* __restrict__ o) {
  int i = blockIdx.x * 256 + threadIdx.x;           // 2097152
  int j = i & 15, lane = (i >> 4) & 31, kt = (i >> 9) & 1, nt = i >> 10;
  int k = 32 * kt + ((j >> 3) << 4) + ((lane >> 4) & 1) * 8 + (j & 7);
  int n = nt * 16 + (lane & 15);
  o[i] = (_Float16)w[k * FLAT + n];
}

// dec_w1 [64,128,4,4] -> per-phase A-operand [ph(4)][mt(4)][kt(16)][lane][16]
__global__ void prep_w1(const float* __restrict__ w, _Float16* __restrict__ o) {
  int i = blockIdx.x * 256 + threadIdx.x;           // 131072
  int j = i & 15, lane = (i >> 4) & 31, kt = (i >> 9) & 15;
  int mt = (i >> 13) & 3, ph = i >> 15;
  int py = ph >> 1, px = ph & 1;
  int k = a_k_of(lane, j) + 32 * kt;                // k = ic*4 + ey*2 + ex
  int ic = k >> 2, ey = (k >> 1) & 1, ex = k & 1;
  int ky = 2 * ey + py, kx = 2 * ex + px;
  int oc = mt * 16 + (lane & 15);
  o[i] = (_Float16)w[oc * 2048 + ic * 16 + ky * 4 + kx];
}

// ======================= conv1 (direct, C_in=3) ============================
__global__ void conv1_direct(const float* __restrict__ x, const float* __restrict__ w,
                             const float* __restrict__ b, _Float16* __restrict__ h1) {
  int i = blockIdx.x * 256 + threadIdx.x;           // B*64*32*32
  int xo = i & 31, yo = (i >> 5) & 31, oc = (i >> 10) & 63, bb = i >> 16;
  float acc = b[oc];
  for (int ic = 0; ic < 3; ++ic)
    for (int ky = 0; ky < 4; ++ky) {
      int yi = 2 * yo - 1 + ky;
      if ((unsigned)yi >= 64u) continue;
      for (int kx = 0; kx < 4; ++kx) {
        int xi = 2 * xo - 1 + kx;
        if ((unsigned)xi >= 64u) continue;
        acc += x[((bb * 3 + ic) * 64 + yi) * 64 + xi] *
               w[((oc * 3 + ic) * 4 + ky) * 4 + kx];
      }
    }
  h1[i] = (_Float16)fmaxf(acc, 0.f);
}

// ================== conv2 as implicit GEMM (WMMA) ==========================
// M=128, K=1024, N=65536. Whole-K B-tile (1024x16 f16 = 32KB) staged once in
// LDS in operand order; one barrier; 32 back-to-back WMMAs per wave.
__global__ void conv2_wmma(const _Float16* __restrict__ h1,   // [B,64,32,32]
                           const _Float16* __restrict__ w2p,
                           const float* __restrict__ bias,
                           _Float16* __restrict__ h2) {       // [B,128,16,16]
  __shared__ _Float16 Bt[32 * 32 * 16];             // [kt][lane][16]
  const int tid = threadIdx.x, lane = tid & 31, wv = tid >> 5;
  const int nbase = blockIdx.x * 16;
  // branchless im2col stage in operand order
  for (int e = tid; e < 16384; e += 256) {
    int j = e & 15, lp = (e >> 4) & 31, kt = e >> 9;
    int k = kt * 32 + ((j >> 3) << 4) + ((lp >> 4) & 1) * 8 + (j & 7);
    int n = nbase + (lp & 15);
    int xo = n & 15, yo = (n >> 4) & 15, bb = n >> 8;
    int ic = k >> 4, ky = (k >> 2) & 3, kx = k & 3;
    int yi = 2 * yo - 1 + ky, xi = 2 * xo - 1 + kx;
    bool inb = ((unsigned)yi < 32u) & ((unsigned)xi < 32u);
    int idx = ((bb * 64 + ic) * 32 + (inb ? yi : 0)) * 32 + (inb ? xi : 0);
    _Float16 raw = h1[idx];
    Bt[e] = inb ? raw : (_Float16)0.f;
  }
  __syncthreads();
  v8f acc = {};
  for (int kt = 0; kt < 32; ++kt) {
    v16h a = *(const v16h*)&w2p[((wv * 32 + kt) * 32 + lane) * 16];
    v16h b = *(const v16h*)&Bt[(kt * 32 + lane) * 16];
    acc = wmma16(a, b, acc);
  }
  int n = nbase + (lane & 15), half = (lane >> 4) & 1;
  int xo = n & 15, yo = (n >> 4) & 15, bb = n >> 8;
#pragma unroll
  for (int r = 0; r < 8; ++r) {
    int oc = wv * 16 + r + 8 * half;                // C layout: M = r + 8*half
    float v = acc[r] + bias[oc];
    h2[((bb * 128 + oc) * 16 + yo) * 16 + xo] = (_Float16)fmaxf(v, 0.f);
  }
}

// =========== encoder linear (WMMA, split-K x16, deterministic) =============
// z_e = h2[256,32768] @ W[32768,64]. 64 tiles x 16 K-slices = 1024 waves.
__global__ void linear_enc_splitk(const _Float16* __restrict__ h2,
                                  const _Float16* __restrict__ wp,
                                  float* __restrict__ zsp) {  // [16][256][64]
  const int tid = threadIdx.x, lane = tid & 31, wv = tid >> 5;
  const int g = blockIdx.x * 8 + wv;                // 0..1023
  const int tile = g >> 4, ks = g & 15;
  const int mt = tile >> 2, nt = tile & 3;
  const int arow = mt * 16 + (lane & 15), half = (lane >> 4) & 1;
  v8f acc = {};
  for (int kt = ks * 64; kt < ks * 64 + 64; ++kt) {
    v16h a = load_a_runs(&h2[arow * FLAT + kt * 32 + half * 8]);
    v16h b = *(const v16h*)&wp[((nt * 1024 + kt) * 32 + lane) * 16];
    acc = wmma16(a, b, acc);
  }
  int n = nt * 16 + (lane & 15);
#pragma unroll
  for (int r = 0; r < 8; ++r) {
    int m = mt * 16 + r + 8 * half;
    zsp[(ks * 256 + m) * 64 + n] = acc[r];
  }
}

__global__ void linear_enc_reduce(const float* __restrict__ zsp,
                                  const float* __restrict__ bias,
                                  float* __restrict__ z_e) {
  int i = blockIdx.x * 256 + threadIdx.x;           // 16384
  int n = i & 63;
  float s = bias[n];
#pragma unroll
  for (int ks = 0; ks < 16; ++ks) s += zsp[ks * 16384 + i];
  z_e[i] = s;
}

// ====================== vector quantization ================================
__global__ void vq_quantize(const float* __restrict__ z_e,
                            const float* __restrict__ cb,
                            float* __restrict__ out_zq, float* __restrict__ out_idx,
                            _Float16* __restrict__ zq_h, float* __restrict__ lossPart) {
  __shared__ float zrow[LATENT];
  __shared__ float bestd[256];
  __shared__ int   besti[256];
  const int b = blockIdx.x, t = threadIdx.x;
  if (t < LATENT) zrow[t] = z_e[b * LATENT + t];
  __syncthreads();
  float bd = 3.4e38f; int bi = 0;
  for (int c = t; c < NCODES; c += 256) {
    float d = 0.f;
    for (int k = 0; k < LATENT; ++k) { float df = zrow[k] - cb[c * LATENT + k]; d += df * df; }
    if (d < bd || (d == bd && c < bi)) { bd = d; bi = c; }
  }
  bestd[t] = bd; besti[t] = bi;
  __syncthreads();
  for (int s = 128; s > 0; s >>= 1) {
    if (t < s) {
      if (bestd[t + s] < bestd[t] ||
          (bestd[t + s] == bestd[t] && besti[t + s] < besti[t])) {
        bestd[t] = bestd[t + s]; besti[t] = besti[t + s];
      }
    }
    __syncthreads();
  }
  const int idx = besti[0];
  if (t == 0) out_idx[b] = (float)idx;
  if (t < LATENT) {
    float q = cb[idx * LATENT + t];
    out_zq[b * LATENT + t] = q;
    zq_h[b * LATENT + t] = (_Float16)q;
    float df = q - zrow[t];
    bestd[t] = df * df;
  }
  __syncthreads();
  if (t == 0) {
    float s = 0.f;
    for (int k = 0; k < LATENT; ++k) s += bestd[k];
    lossPart[b] = s;                                // deterministic partial
  }
}

__global__ void vq_finalize(const float* __restrict__ lossPart,
                            float* __restrict__ l1, float* __restrict__ l2) {
  __shared__ float red[256];
  int t = threadIdx.x;
  red[t] = lossPart[t];
  __syncthreads();
  for (int s = 128; s > 0; s >>= 1) {
    if (t < s) red[t] += red[t + s];
    __syncthreads();
  }
  if (t == 0) {
    float v = red[0] / (float)(BATCH * LATENT);
    l1[0] = v; l2[0] = v;                           // vq == commitment in fwd
  }
}

// ================== decoder linear (WMMA): d = relu(z_q @ W + b) ===========
__global__ void linear_dec_wmma(const _Float16* __restrict__ zq,  // [256,64]
                                const _Float16* __restrict__ wp,
                                const float* __restrict__ bias,
                                _Float16* __restrict__ dlin) {    // [256,32768]
  const int tid = threadIdx.x, lane = tid & 31, wv = tid >> 5;
  const int g = blockIdx.x * 8 + wv;                // 32768 tiles
  const int nt = g & 2047, mt = g >> 11;
  const int arow = mt * 16 + (lane & 15), half = (lane >> 4) & 1;
  v8f acc = {};
#pragma unroll
  for (int kt = 0; kt < 2; ++kt) {
    v16h a = load_a_runs(&zq[arow * LATENT + kt * 32 + half * 8]);
    v16h b = *(const v16h*)&wp[((nt * 2 + kt) * 32 + lane) * 16];
    acc = wmma16(a, b, acc);
  }
  int n = nt * 16 + (lane & 15);
#pragma unroll
  for (int r = 0; r < 8; ++r) {
    int bb = mt * 16 + r + 8 * half;
    float v = acc[r] + bias[n];
    dlin[bb * FLAT + n] = (_Float16)fmaxf(v, 0.f);
  }
}

// ============== transposed conv1 as phase-split implicit GEMM ==============
// Per phase (py,px): K = ic(128)*ey(2)*ex(2) = 512. Whole-K B-tile for 32
// output columns (512x32 f16 = 32KB) staged once; 8 waves = 4 Mt x 2 sub.
__global__ void convt1_wmma(const _Float16* __restrict__ dlin,  // [B,128,16,16]
                            const _Float16* __restrict__ w1p,
                            const float* __restrict__ bias,
                            _Float16* __restrict__ d1) {        // [B,64,32,32]
  __shared__ _Float16 Bt[16 * 2 * 32 * 16];         // [kt][sub][lane][16]
  const int tid = threadIdx.x, lane = tid & 31, wv = tid >> 5;
  const int mt = wv & 3, sub = wv >> 2;
  const int ph = blockIdx.y, py = ph >> 1, px = ph & 1;
  const int nbase = blockIdx.x * 32;
  for (int e = tid; e < 16384; e += 256) {
    int j = e & 15, lp = (e >> 4) & 31, sb = (e >> 9) & 1, kt = e >> 10;
    int k = kt * 32 + ((j >> 3) << 4) + ((lp >> 4) & 1) * 8 + (j & 7);     // ic*4 + ey*2 + ex
    int n = nbase + sb * 16 + (lp & 15);
    int xx = n & 15, yy = (n >> 4) & 15, bb = n >> 8;
    int ic = k >> 2, ey = (k >> 1) & 1, ex = k & 1;
    int ky = 2 * ey + py, kx = 2 * ex + px;
    int iy = yy + ((py + ky - 2) >> 1);
    int ix = xx + ((px + kx - 2) >> 1);
    bool inb = ((unsigned)iy < 16u) & ((unsigned)ix < 16u);
    int idx = ((bb * 128 + ic) * 16 + (inb ? iy : 0)) * 16 + (inb ? ix : 0);
    _Float16 raw = dlin[idx];
    Bt[e] = inb ? raw : (_Float16)0.f;
  }
  __syncthreads();
  v8f acc = {};
  for (int kt = 0; kt < 16; ++kt) {
    v16h a = *(const v16h*)&w1p[(((ph * 4 + mt) * 16 + kt) * 32 + lane) * 16];
    v16h b = *(const v16h*)&Bt[((kt * 2 + sub) * 32 + lane) * 16];
    acc = wmma16(a, b, acc);
  }
  int n = nbase + sub * 16 + (lane & 15), half = (lane >> 4) & 1;
  int xx = n & 15, yy = (n >> 4) & 15, bb = n >> 8;
  int y = 2 * yy + py, x = 2 * xx + px;
#pragma unroll
  for (int r = 0; r < 8; ++r) {
    int oc = mt * 16 + r + 8 * half;
    float v = acc[r] + bias[oc];
    d1[((bb * 64 + oc) * 32 + y) * 32 + x] = (_Float16)fmaxf(v, 0.f);
  }
}

// ============== transposed conv2 (direct, C_out=3) + sigmoid ===============
__global__ void convt2_sigmoid(const _Float16* __restrict__ d1,  // [B,64,32,32]
                               const float* __restrict__ w,      // [3,64,4,4]
                               const float* __restrict__ bias,
                               float* __restrict__ out) {        // [B,3,64,64]
  int i = blockIdx.x * 256 + threadIdx.x;           // 3145728
  int x = i & 63, y = (i >> 6) & 63;
  int rest = i >> 12;
  int oc = rest % 3, bb = rest / 3;
  int py = y & 1, px = x & 1, yy = y >> 1, xx = x >> 1;
  float acc = bias[oc];
  for (int ic = 0; ic < 64; ++ic)
    for (int ey = 0; ey < 2; ++ey) {
      int ky = 2 * ey + py, iy = yy + ((py + ky - 2) >> 1);
      if ((unsigned)iy >= 32u) continue;
      for (int ex = 0; ex < 2; ++ex) {
        int kx = 2 * ex + px, ix = xx + ((px + kx - 2) >> 1);
        if ((unsigned)ix >= 32u) continue;
        acc += (float)d1[((bb * 64 + ic) * 32 + iy) * 32 + ix] *
               w[(oc * 64 + ic) * 16 + ky * 4 + kx];
      }
    }
  out[i] = 1.f / (1.f + __expf(-acc));
}

// =============================== launch ====================================
extern "C" void kernel_launch(void* const* d_in, const int* in_sizes, int n_in,
                              void* d_out, int out_size, void* d_ws, size_t ws_size,
                              hipStream_t stream) {
  const float* x      = (const float*)d_in[0];
  const float* enc_w1 = (const float*)d_in[1];
  const float* enc_b1 = (const float*)d_in[2];
  const float* enc_w2 = (const float*)d_in[3];
  const float* enc_b2 = (const float*)d_in[4];
  const float* enc_wl = (const float*)d_in[5];
  const float* enc_bl = (const float*)d_in[6];
  const float* cb     = (const float*)d_in[7];
  const float* dec_wl = (const float*)d_in[8];
  const float* dec_bl = (const float*)d_in[9];
  const float* dec_w1 = (const float*)d_in[10];
  const float* dec_b1 = (const float*)d_in[11];
  const float* dec_w2 = (const float*)d_in[12];
  const float* dec_b2 = (const float*)d_in[13];

  char* ws = (char*)d_ws;
  size_t off = 0;
  auto take = [&](size_t bytes) { char* p = ws + off; off = (off + bytes + 255) & ~(size_t)255; return p; };
  _Float16* w2p   = (_Float16*)take(131072  * 2);
  _Float16* ewlp  = (_Float16*)take(2097152 * 2);
  _Float16* dwlp  = (_Float16*)take(2097152 * 2);
  _Float16* w1p   = (_Float16*)take(131072  * 2);
  _Float16* h1    = (_Float16*)take(16777216u * 2); // [B,64,32,32]
  _Float16* h2    = (_Float16*)take(8388608u  * 2); // [B,128,16,16]
  float*    z_e   = (float*)   take(16384 * 4);
  _Float16* zq_h  = (_Float16*)take(16384 * 2);
  _Float16* dlin  = (_Float16*)take(8388608u * 2);
  _Float16* d1    = (_Float16*)take(16777216u * 2);
  float*    zsp   = (float*)   take(16 * 16384 * 4); // split-K partials
  float*    lossP = (float*)   take(256 * 4);
  (void)ws_size; (void)in_sizes; (void)n_in; (void)out_size;

  float* o_rec = (float*)d_out;
  float* o_zq  = o_rec + 3145728;
  float* o_vql = o_zq + 16384;
  float* o_cl  = o_vql + 1;
  float* o_idx = o_cl + 1;

  prep_w2   <<<512,  256, 0, stream>>>(enc_w2, w2p);
  prep_encwl<<<8192, 256, 0, stream>>>(enc_wl, ewlp);
  prep_decwl<<<8192, 256, 0, stream>>>(dec_wl, dwlp);
  prep_w1   <<<512,  256, 0, stream>>>(dec_w1, w1p);

  conv1_direct     <<<65536, 256, 0, stream>>>(x, enc_w1, enc_b1, h1);
  conv2_wmma       <<<4096,  256, 0, stream>>>(h1, w2p, enc_b2, h2);
  linear_enc_splitk<<<128,   256, 0, stream>>>(h2, ewlp, zsp);
  linear_enc_reduce<<<64,    256, 0, stream>>>(zsp, enc_bl, z_e);
  vq_quantize      <<<256,   256, 0, stream>>>(z_e, cb, o_zq, o_idx, zq_h, lossP);
  vq_finalize      <<<1,     256, 0, stream>>>(lossP, o_vql, o_cl);
  linear_dec_wmma  <<<4096,  256, 0, stream>>>(zq_h, dwlp, dec_bl, dlin);
  convt1_wmma      <<<dim3(2048, 4), 256, 0, stream>>>(dlin, w1p, dec_b1, d1);
  convt2_sigmoid   <<<12288, 256, 0, stream>>>(d1, dec_w2, dec_b2, o_rec);
}
